// HGT_80376017977984
// MI455X (gfx1250) — matrix operand order repaired and checked by reading, not compiled, not run on hardware
//
#include <hip/hip_runtime.h>
#include <hip/hip_bf16.h>
#include <cmath>

#define HID    128
#define HEADS  8
#define DHEAD  16

typedef __attribute__((ext_vector_type(16))) __bf16 bf16x16;
typedef __attribute__((ext_vector_type(8)))  float  f32x8;

union FragBF {
  bf16x16      v;
  unsigned int u32[8];
};

__device__ __forceinline__ unsigned short f32_bf16(float f) {
  unsigned int u = __float_as_uint(f);
  u += 0x7fffu + ((u >> 16) & 1u);   // round to nearest even
  return (unsigned short)(u >> 16);
}

__device__ __forceinline__ unsigned int pack_bf16x2(float lo, float hi) {
  return (unsigned int)f32_bf16(lo) | ((unsigned int)f32_bf16(hi) << 16);
}

__device__ __forceinline__ float gelu_exact(float x) {
  return 0.5f * x * (1.f + erff(x * 0.70710678118f));
}

// ---------------------------------------------------------------------------
// WMMA GEMM: C[M,128] = epilogue( A[M,K] @ B[K,128] + bias )
//   preGelu : exact GELU applied to A elements while staging
//   postRelu: ReLU on result
//   skipP   : if non-null, C = sigmoid(*skipP)*C + (1-sig)*xOld
// Block = 256 threads (8 wave32), tile 64x128, K-step 64 (2 WMMA sub-steps),
// bf16 WMMA with f32 accumulation. Fast path: unguarded b128 staging.
// ---------------------------------------------------------------------------
__global__ __launch_bounds__(256)
void hgt_gemm128(const float* __restrict__ A, int M, int K,
                 const float* __restrict__ B, const float* __restrict__ bias,
                 float* __restrict__ C,
                 int preGelu, int postRelu,
                 const float* __restrict__ skipP, const float* __restrict__ xOld)
{
  constexpr int AP = 68;                  // pitch in halves: 136B rows, 8B aligned
  __shared__ unsigned short As[64][AP];   // [m][k]
  __shared__ unsigned short Bs[128][AP];  // [n][k] (transposed)

  const int t     = threadIdx.x;
  const int wave  = t >> 5;
  const int lane  = t & 31;
  const int m0    = blockIdx.x * 64;
  const int mbase = (wave >> 1) << 4;     // 0,16,32,48
  const int nbase = (wave & 1) << 6;      // 0 or 64
  const int tr    = lane & 15;
  const int hi    = lane >> 4;            // lane group 0/1
  const bool mFull = (m0 + 64 <= M);

  f32x8 acc[4];
  for (int s = 0; s < 4; ++s)
    for (int i = 0; i < 8; ++i) acc[s][i] = 0.f;

  const int nkt = (K + 63) >> 6;
  for (int kt = 0; kt < nkt; ++kt) {
    const int k0 = kt << 6;
    const bool full = mFull && (k0 + 64 <= K);
    __syncthreads();
    if (full) {
      // ---- fast path: unguarded float4 staging ----
      // A tile 64x64 floats = 1024 float4, 4 per thread, coalesced
      #pragma unroll
      for (int i = 0; i < 4; ++i) {
        int idx = t + (i << 8);
        int r = idx >> 4, c4 = idx & 15;
        float4 v = *(const float4*)(A + (long)(m0 + r) * K + k0 + (c4 << 2));
        if (preGelu) {
          v.x = gelu_exact(v.x); v.y = gelu_exact(v.y);
          v.z = gelu_exact(v.z); v.w = gelu_exact(v.w);
        }
        unsigned int* d = (unsigned int*)&As[r][c4 << 2];
        d[0] = pack_bf16x2(v.x, v.y);
        d[1] = pack_bf16x2(v.z, v.w);
      }
      // B tile 64x128 floats = 2048 float4, 8 per thread, coalesced
      #pragma unroll
      for (int i = 0; i < 8; ++i) {
        int idx = t + (i << 8);
        int kr = idx >> 5, n4 = idx & 31;
        float4 v = *(const float4*)(B + (long)(k0 + kr) * HID + (n4 << 2));
        int n = n4 << 2;
        Bs[n + 0][kr] = f32_bf16(v.x);
        Bs[n + 1][kr] = f32_bf16(v.y);
        Bs[n + 2][kr] = f32_bf16(v.z);
        Bs[n + 3][kr] = f32_bf16(v.w);
      }
    } else {
      // ---- guarded path: branchless (clamped address + select) ----
      #pragma unroll
      for (int i = 0; i < 16; ++i) {
        int idx = t + (i << 8);
        int r = idx >> 6, c = idx & 63;
        int gr = m0 + r, gk = k0 + c;
        int grc = gr < M ? gr : M - 1;
        int gkc = gk < K ? gk : K - 1;
        float x = A[(long)grc * K + gkc];
        x = (gr < M && gk < K) ? x : 0.f;
        if (preGelu) x = gelu_exact(x);
        As[r][c] = f32_bf16(x);
      }
      #pragma unroll
      for (int i = 0; i < 32; ++i) {
        int idx = t + (i << 8);
        int kr = idx >> 7, n = idx & 127;
        int gk = k0 + kr;
        int gkc = gk < K ? gk : K - 1;
        float x = B[(long)gkc * HID + n];
        x = (gk < K) ? x : 0.f;
        Bs[n][kr] = f32_bf16(x);
      }
    }
    __syncthreads();

    // two 16x16x32 sub-steps over the 64-wide K stage
    #pragma unroll
    for (int kh = 0; kh < 2; ++kh) {
      // A fragment: lane<16 holds K 0..7 & 16..23, lane>=16 K 8..15 & 24..31 (per sub-step)
      FragBF a;
      {
        const unsigned int* ap = (const unsigned int*)(&As[mbase + tr][0]);
        const int ko = (kh << 4) + (hi << 2);
        #pragma unroll
        for (int j = 0; j < 4; ++j) {
          a.u32[j]     = ap[ko + j];
          a.u32[4 + j] = ap[ko + 8 + j];
        }
      }
      #pragma unroll
      for (int s = 0; s < 4; ++s) {
        FragBF b;
        const unsigned int* bp = (const unsigned int*)(&Bs[nbase + (s << 4) + tr][0]);
        const int kb = (kh << 4) + (hi << 2);
        #pragma unroll
        for (int j = 0; j < 8; ++j) b.u32[j] = bp[kb + j + ((j >> 2) << 2)];
        acc[s] = __builtin_amdgcn_wmma_f32_16x16x32_bf16(
            false, a.v, false, b.v, (short)0, acc[s], false, false);
      }
    }
  }

  float sk = 0.f, osk = 0.f;
  if (skipP) { float sv = skipP[0]; sk = 1.f / (1.f + expf(-sv)); osk = 1.f - sk; }

  // D layout: VGPR r -> row mbase + r + 8*hi, col = nbase + tr
  #pragma unroll
  for (int s = 0; s < 4; ++s) {
    const int colG = nbase + (s << 4) + tr;
    #pragma unroll
    for (int r = 0; r < 8; ++r) {
      int rowG = m0 + mbase + r + (hi << 3);
      if (rowG < M) {
        float c = acc[s][r] + bias[colG];
        if (postRelu) c = fmaxf(c, 0.f);
        if (skipP) c = sk * c + osk * xOld[(long)rowG * HID + colG];
        C[(long)rowG * HID + colG] = c;
      }
    }
  }
}

// ---------------------------------------------------------------------------
// Fold per-head relation matrix into projection weights:
//   Wf[k, h*16+e] = sum_d W[k, h*16+d] * rel[h, d, e]
// ---------------------------------------------------------------------------
__global__ void hgt_fuse_w(const float* __restrict__ W, const float* __restrict__ rel,
                           float* __restrict__ Wf)
{
  int idx = blockIdx.x * 256 + threadIdx.x;
  if (idx >= HID * HID) return;
  int k = idx >> 7, n = idx & 127;
  int h = n >> 4, e = n & 15;
  float s = 0.f;
  #pragma unroll
  for (int d = 0; d < 16; ++d)
    s += W[k * HID + (h << 4) + d] * rel[h * 256 + d * 16 + e];
  Wf[k * HID + n] = s;
}

__global__ void hgt_fuse_b(const float* __restrict__ b, const float* __restrict__ rel,
                           float* __restrict__ bf)
{
  int n = threadIdx.x;  // 128
  int h = n >> 4, e = n & 15;
  float s = 0.f;
  #pragma unroll
  for (int d = 0; d < 16; ++d)
    s += b[(h << 4) + d] * rel[h * 256 + d * 16 + e];
  bf[n] = s;
}

// ---------------------------------------------------------------------------
// Edge phase
// ---------------------------------------------------------------------------
__device__ __forceinline__ void atomicMaxF(float* addr, float v) {
  if (v >= 0.f) atomicMax((int*)addr, __float_as_int(v));
  else          atomicMin((unsigned int*)addr, __float_as_uint(v));
}

__global__ void hgt_fill(float* __restrict__ p, int n, float v) {
  int i = blockIdx.x * 256 + threadIdx.x;
  if (i < n) p[i] = v;
}

__global__ void hgt_edge_alpha(const int* __restrict__ src, const int* __restrict__ dst, int E,
                               const float* __restrict__ q, const float* __restrict__ krel,
                               const float* __restrict__ prel,
                               float* __restrict__ alpha, float* __restrict__ amax)
{
  int idx = blockIdx.x * 256 + threadIdx.x;
  if (idx >= E * HEADS) return;
  int e = idx >> 3, h = idx & 7;
  int s = src[e], d = dst[e];
  const float4* qp = (const float4*)(q    + (long)d * HID + h * DHEAD);
  const float4* kp = (const float4*)(krel + (long)s * HID + h * DHEAD);
  float acc = 0.f;
  #pragma unroll
  for (int i = 0; i < 4; ++i) {
    float4 a = qp[i], b = kp[i];
    acc += a.x * b.x + a.y * b.y + a.z * b.z + a.w * b.w;
  }
  float al = acc * prel[h] * 0.25f;  // 1/sqrt(16)
  alpha[(long)e * HEADS + h] = al;
  atomicMaxF(&amax[d * HEADS + h], al);
}

__global__ void hgt_edge_exp(const int* __restrict__ dst, int E,
                             float* __restrict__ alpha, const float* __restrict__ amax,
                             float* __restrict__ den)
{
  int idx = blockIdx.x * 256 + threadIdx.x;
  if (idx >= E * HEADS) return;
  int e = idx >> 3, h = idx & 7;
  int d = dst[e];
  float ex = expf(alpha[(long)e * HEADS + h] - amax[d * HEADS + h]);
  alpha[(long)e * HEADS + h] = ex;
  atomicAdd(&den[d * HEADS + h], ex);
}

__global__ void hgt_edge_scatter(const int* __restrict__ src, const int* __restrict__ dst, int E,
                                 const float* __restrict__ vrel, const float* __restrict__ alpha,
                                 const float* __restrict__ den, float* __restrict__ out)
{
  int idx = blockIdx.x * 256 + threadIdx.x;
  if (idx >= E * HEADS) return;
  int e = idx >> 3, h = idx & 7;
  int s = src[e], d = dst[e];
  float w = alpha[(long)e * HEADS + h] / (den[d * HEADS + h] + 1e-16f);
  const float4* vp = (const float4*)(vrel + (long)s * HID + h * DHEAD);
  float* op = out + (long)d * HID + h * DHEAD;
  #pragma unroll
  for (int i = 0; i < 4; ++i) {
    float4 v = vp[i];
    atomicAdd(op + 4 * i + 0, v.x * w);
    atomicAdd(op + 4 * i + 1, v.y * w);
    atomicAdd(op + 4 * i + 2, v.z * w);
    atomicAdd(op + 4 * i + 3, v.w * w);
  }
}

// ---------------------------------------------------------------------------
// Output head: y = sigmoid(xd @ W[128,1] + b)
// ---------------------------------------------------------------------------
__global__ void hgt_head(const float* __restrict__ xd, int N,
                         const float* __restrict__ W, const float* __restrict__ b,
                         float* __restrict__ y)
{
  int n = blockIdx.x * 256 + threadIdx.x;
  if (n >= N) return;
  const float4* xp = (const float4*)(xd + (long)n * HID);
  const float4* wp = (const float4*)W;
  float acc = 0.f;
  #pragma unroll
  for (int i = 0; i < 32; ++i) {
    float4 a = xp[i], w = wp[i];
    acc += a.x * w.x + a.y * w.y + a.z * w.z + a.w * w.w;
  }
  y[n] = 1.f / (1.f + expf(-(acc + b[0])));
}

// ---------------------------------------------------------------------------
extern "C" void kernel_launch(void* const* d_in, const int* in_sizes, int n_in,
                              void* d_out, int out_size, void* d_ws, size_t ws_size,
                              hipStream_t stream)
{
  (void)in_sizes; (void)n_in; (void)out_size; (void)ws_size;

  // types: 0=drug 1=disease 2=protein
  const int NT[3]  = {30000, 15000, 40000};
  const int ET[5]  = {300000, 300000, 250000, 250000, 400000}; // dd,dp,drdr,pd,pp
  const int EST[5] = {0, 0, 0, 2, 2};
  const int EDT[5] = {1, 2, 0, 1, 2};
  const int SRT[3] = {1, 0, 2};   // type idx -> sorted-key idx (disease,drug,protein)

  auto F = [&](int i) -> const float* { return (const float*)d_in[i]; };
  auto I = [&](int i) -> const int*   { return (const int*)d_in[i]; };

  // params flattened (jax pytree, sorted dict keys) after 3 x + 10 edge arrays
  const int PI = 13;
  auto LA    = [&](int l, int t) { return PI + 42 * l + 0  + 2 * SRT[t]; };
  auto LAREL = [&](int l, int e) { return PI + 42 * l + 6  + e; };
  auto LK    = [&](int l, int t) { return PI + 42 * l + 11 + 2 * SRT[t]; };
  auto LMREL = [&](int l, int e) { return PI + 42 * l + 17 + e; };
  auto LPREL = [&](int l, int e) { return PI + 42 * l + 22 + e; };
  auto LQ    = [&](int l, int t) { return PI + 42 * l + 27 + 2 * SRT[t]; };
  auto LSKIP = [&](int l, int t) { return PI + 42 * l + 33 + SRT[t]; };
  auto LV    = [&](int l, int t) { return PI + 42 * l + 36 + 2 * SRT[t]; };
  const int LIN = PI + 84, OUTW = PI + 90, OUTB = PI + 91;

  // workspace bump allocator
  char* wsp = (char*)d_ws;
  size_t off = 0;
  auto alloc = [&](size_t bytes) -> float* {
    float* p = (float*)(wsp + off);
    off = (off + bytes + 255) & ~(size_t)255;
    return p;
  };
  float *xdA[3], *xdB[3], *qb[3], *outg[3];
  for (int t = 0; t < 3; ++t) xdA[t]  = alloc((size_t)NT[t] * HID * 4);
  for (int t = 0; t < 3; ++t) xdB[t]  = alloc((size_t)NT[t] * HID * 4);
  for (int t = 0; t < 3; ++t) qb[t]   = alloc((size_t)NT[t] * HID * 4);
  for (int t = 0; t < 3; ++t) outg[t] = alloc((size_t)NT[t] * HID * 4);
  float* krel  = alloc((size_t)40000 * HID * 4);
  float* vrel  = alloc((size_t)40000 * HID * 4);
  float* alpha = alloc((size_t)400000 * HEADS * 4);
  float* amax  = alloc((size_t)40000 * HEADS * 4);
  float* den   = alloc((size_t)40000 * HEADS * 4);
  float* Wfk   = alloc(HID * HID * 4);
  float* bfk   = alloc(HID * 4);
  float* Wfv   = alloc(HID * HID * 4);
  float* bfv   = alloc(HID * 4);

  auto gemm = [&](const float* A, int M, int K, const float* B, const float* bias,
                  float* C, int pg, int pr, const float* sp, const float* xo) {
    hgt_gemm128<<<dim3((M + 63) / 64), dim3(256), 0, stream>>>(A, M, K, B, bias, C, pg, pr, sp, xo);
  };
  auto fill = [&](float* p, int n, float v) {
    hgt_fill<<<dim3((n + 255) / 256), dim3(256), 0, stream>>>(p, n, v);
  };

  // input linears + ReLU
  for (int t = 0; t < 3; ++t)
    gemm(F(t), NT[t], 732, F(LIN + 2 * SRT[t]), F(LIN + 2 * SRT[t] + 1),
         xdA[t], 0, 1, nullptr, nullptr);

  float* cur[3] = {xdA[0], xdA[1], xdA[2]};
  float* nxt[3] = {xdB[0], xdB[1], xdB[2]};

  for (int l = 0; l < 2; ++l) {
    // Q projections
    for (int t = 0; t < 3; ++t)
      gemm(cur[t], NT[t], HID, F(LQ(l, t)), F(LQ(l, t) + 1), qb[t], 0, 0, nullptr, nullptr);
    // zero message aggregators
    for (int t = 0; t < 3; ++t) fill(outg[t], NT[t] * HID, 0.f);

    for (int e = 0; e < 5; ++e) {
      int st = EST[e], dt = EDT[e], E = ET[e];
      // fold a_rel/m_rel into K/V weights
      hgt_fuse_w<<<dim3(64), dim3(256), 0, stream>>>(F(LK(l, st)), F(LAREL(l, e)), Wfk);
      hgt_fuse_b<<<dim3(1), dim3(128), 0, stream>>>(F(LK(l, st) + 1), F(LAREL(l, e)), bfk);
      hgt_fuse_w<<<dim3(64), dim3(256), 0, stream>>>(F(LV(l, st)), F(LMREL(l, e)), Wfv);
      hgt_fuse_b<<<dim3(1), dim3(128), 0, stream>>>(F(LV(l, st) + 1), F(LMREL(l, e)), bfv);
      // k_rel / v_rel as full 128x128 WMMA GEMMs
      gemm(cur[st], NT[st], HID, Wfk, bfk, krel, 0, 0, nullptr, nullptr);
      gemm(cur[st], NT[st], HID, Wfv, bfv, vrel, 0, 0, nullptr, nullptr);
      // segment softmax (3 passes) + scatter
      fill(amax, NT[dt] * HEADS, -INFINITY);
      fill(den,  NT[dt] * HEADS, 0.f);
      int g = (E * HEADS + 255) / 256;
      hgt_edge_alpha<<<dim3(g), dim3(256), 0, stream>>>(
          I(3 + 2 * e), I(4 + 2 * e), E, qb[dt], krel, F(LPREL(l, e)), alpha, amax);
      hgt_edge_exp<<<dim3(g), dim3(256), 0, stream>>>(I(4 + 2 * e), E, alpha, amax, den);
      hgt_edge_scatter<<<dim3(g), dim3(256), 0, stream>>>(
          I(3 + 2 * e), I(4 + 2 * e), E, vrel, alpha, den, outg[dt]);
    }

    // new = sigmoid(skip)*(GELU(out) @ Wa + ba) + (1-sig)*xd  (fused epilogue)
    for (int t = 0; t < 3; ++t)
      gemm(outg[t], NT[t], HID, F(LA(l, t)), F(LA(l, t) + 1), nxt[t],
           1, 0, F(LSKIP(l, t)), cur[t]);
    for (int t = 0; t < 3; ++t) { float* tmp = cur[t]; cur[t] = nxt[t]; nxt[t] = tmp; }
  }

  // output heads: drug then protein, concatenated
  float* yout = (float*)d_out;
  hgt_head<<<dim3((NT[0] + 255) / 256), dim3(256), 0, stream>>>(
      cur[0], NT[0], F(OUTW), F(OUTB), yout);
  hgt_head<<<dim3((NT[2] + 255) / 256), dim3(256), 0, stream>>>(
      cur[2], NT[2], F(OUTW), F(OUTB), yout + NT[0]);
}
